// CombinedBandPassFilter_4896262717873
// MI455X (gfx1250) — matrix operand order repaired and checked by reading, not compiled, not run on hardware
//
#include <hip/hip_runtime.h>
#include <hip/hip_bf16.h>
#include <stdint.h>

typedef __bf16 bf16;
typedef __attribute__((ext_vector_type(16))) __bf16 v16bf;
typedef __attribute__((ext_vector_type(8)))  float  v8f;
typedef uint32_t u32x4 __attribute__((ext_vector_type(4)));
typedef int      i32x4 __attribute__((ext_vector_type(4)));
typedef int      i32x8 __attribute__((ext_vector_type(8)));

#define T_LEN   32768
#define NBATCH  32
#define NFILT   40
#define KTAPS   769
#define FPAD    384          // (769-1)/2, 'same' padding
#define KSTEPS  25           // K padded to 25*32 = 800
#define MTILES  3            // 48 filter rows (40 padded to 48)
#define TILE_T  256          // time outputs per block
#define WIN     1056         // 256 + 800 window (max element used = 1054)
#define WIND    (WIN / 2)    // 528 dwords (bf16 pairs)
#define AFRAG_TERM (MTILES * KSTEPS * 32 * 16)   // 38400 bf16 per term

// ---------------------------------------------------------------------------
// Prep: swizzle the 40x769 fp32 filter bank into WMMA A-fragment layout,
// split into bf16 hi / lo terms (bf16x3 fp32-emulation scheme).
// A-layout (contiguous k, 16-bit A 16x32): lane<16: M=lane, K in {0..15};
// lane>=16: M=lane-16, K in {16..31}; VGPR v packs K={16*half+2v, 16*half+2v+1}.
// ---------------------------------------------------------------------------
__global__ __launch_bounds__(256) void prep_afrag(const float* __restrict__ kern,
                                                  bf16* __restrict__ afrag) {
    int tid = blockIdx.x * blockDim.x + threadIdx.x;
    if (tid >= 2 * AFRAG_TERM) return;
    int term = tid / AFRAG_TERM;
    int r    = tid % AFRAG_TERM;          // [m][s][lane][j]
    int j    = r & 15;
    int l    = (r >> 4) & 31;
    int rest = r >> 9;
    int s    = rest % KSTEPS;
    int m    = rest / KSTEPS;

    int half = l >> 4;
    int M    = l & 15;
    int v    = j >> 1, p = j & 1;
    int K    = half * 16 + v * 2 + p;

    int f  = m * 16 + M;
    int kk = s * 32 + K;
    float val = (f < NFILT && kk < KTAPS) ? kern[f * KTAPS + kk] : 0.0f;
    bf16 hi = (bf16)val;
    afrag[tid] = (term == 0) ? hi : (bf16)(val - (float)hi);
}

static __device__ __forceinline__ uint32_t pack_bf16(bf16 a, bf16 b) {
    return (uint32_t)__builtin_bit_cast(uint16_t, a) |
           ((uint32_t)__builtin_bit_cast(uint16_t, b) << 16);
}

// ---------------------------------------------------------------------------
// Implicit-GEMM FIR bank via v_wmma_f32_16x16x32_bf16, bf16x3 split precision.
// Block = 256 threads (8 waves). Each block: one batch x 256 time outputs.
// x window staged raw into LDS by the Tensor Data Mover, then split to
// bf16 hi/lo packed-dword arrays for aligned ds_load_b32 fragment builds.
// ---------------------------------------------------------------------------
__global__ __launch_bounds__(256) void conv_wmma(const float* __restrict__ x,
                                                 const bf16* __restrict__ afrag,
                                                 float* __restrict__ out) {
    __shared__ float    xraw[WIN];
    __shared__ uint32_t xh32[WIND];
    __shared__ uint32_t xl32[WIND];

    const int b   = blockIdx.y;
    const int t0  = blockIdx.x * TILE_T;
    const int tid = threadIdx.x;
    const int wave = tid >> 5;
    const int lane = tid & 31;

    // Zero the raw window (covers edge-block padding), cheap for interior.
    for (int i = tid; i < WIN; i += 256) xraw[i] = 0.0f;
    __syncthreads();

    // --- TDM: DMA the (clamped) f32 window global -> LDS --------------------
    if (wave == 0) {
        const int start = t0 - FPAD;
        const int gLo   = start < 0 ? 0 : start;
        const int gHi   = (start + WIN) > T_LEN ? T_LEN : (start + WIN);
        const int len   = gHi - gLo;                       // <= 1056
        const uint64_t ga = (uint64_t)(uintptr_t)(x + (size_t)b * T_LEN + gLo);
        const uint32_t ldsa = (uint32_t)(uintptr_t)&xraw[0] +
                              (uint32_t)(gLo - start) * 4u;

        u32x4 g0;
        g0[0] = 1u;                                  // count=1, normal mode
        g0[1] = ldsa;                                // lds_addr (bytes)
        g0[2] = (uint32_t)ga;                        // global_addr[31:0]
        g0[3] = (uint32_t)(ga >> 32) | 0x80000000u;  // global_addr[56:32] | type=2

        i32x8 g1;
        g1[0] = 0x20000;                 // workgroup_mask=0, data_size=2 (4B)
        g1[1] = (len & 0xffff) << 16;    // tensor_dim0[15:0]
        g1[2] = (len >> 16) | (1 << 16); // tensor_dim0[31:16] | tensor_dim1=1
        g1[3] = len << 16;               // tile_dim0 = len
        g1[4] = 1;                       // tile_dim1 = 1
        g1[5] = len;                     // tensor_dim0_stride
        g1[6] = 0;
        g1[7] = 0;

        i32x4 gz = (i32x4){0, 0, 0, 0};
#if defined(__clang_major__) && (__clang_major__ >= 23)
        i32x8 gz8 = (i32x8){0, 0, 0, 0, 0, 0, 0, 0};
        __builtin_amdgcn_tensor_load_to_lds(g0, g1, gz, gz, gz8, 0);
#else
        __builtin_amdgcn_tensor_load_to_lds(g0, g1, gz, gz, 0);
#endif
        __builtin_amdgcn_s_wait_tensorcnt(0);
    }
    __syncthreads();

    // --- Split to bf16 hi/lo, packed 2 samples per dword --------------------
    for (int d = tid; d < WIND; d += 256) {
        float v0 = xraw[2 * d], v1 = xraw[2 * d + 1];
        bf16 h0 = (bf16)v0, h1 = (bf16)v1;
        bf16 l0 = (bf16)(v0 - (float)h0), l1 = (bf16)(v1 - (float)h1);
        xh32[d] = pack_bf16(h0, h1);
        xl32[d] = pack_bf16(l0, l1);
    }
    __syncthreads();

    const int half = lane >> 4;
    const int nl   = lane & 15;

    v8f acc[MTILES][2];
    #pragma unroll
    for (int m = 0; m < MTILES; ++m) {
        acc[m][0] = (v8f){0.f,0.f,0.f,0.f,0.f,0.f,0.f,0.f};
        acc[m][1] = (v8f){0.f,0.f,0.f,0.f,0.f,0.f,0.f,0.f};
    }

    // B element (k,n): xwin[ncol + k]; ncol = wave*32 + nt*16 + nl,
    // k = s*32 + half*16 + j  (lane-half holds K 0..15 / 16..31)
    const int nbase = wave * 32 + nl + half * 16;
    const uint32_t sh = (uint32_t)(nbase & 1) << 4;   // funnel shift, per-lane

    union BFrag { v16bf v; uint32_t u[8]; };

    #pragma unroll 1
    for (int s = 0; s < KSTEPS; ++s) {
        const int base0 = nbase + s * 32;
        const int widx  = base0 >> 1;     // aligned dword index into xh32/xl32

        uint32_t wh[17], wl[17];
        #pragma unroll
        for (int i = 0; i < 17; ++i) { wh[i] = xh32[widx + i]; wl[i] = xl32[widx + i]; }

        BFrag bh0, bl0, bh1, bl1;
        #pragma unroll
        for (int j = 0; j < 8; ++j) {
            bh0.u[j] = (uint32_t)((((uint64_t)wh[j + 1] << 32) | wh[j]) >> sh);
            bl0.u[j] = (uint32_t)((((uint64_t)wl[j + 1] << 32) | wl[j]) >> sh);
            bh1.u[j] = (uint32_t)((((uint64_t)wh[j + 9] << 32) | wh[j + 8]) >> sh);
            bl1.u[j] = (uint32_t)((((uint64_t)wl[j + 9] << 32) | wl[j + 8]) >> sh);
        }

        #pragma unroll
        for (int m = 0; m < MTILES; ++m) {
            size_t aoff = ((size_t)((m * KSTEPS + s) * 32 + lane)) * 16;
            v16bf ah = *(const v16bf*)(afrag + aoff);
            v16bf al = *(const v16bf*)(afrag + AFRAG_TERM + aoff);

            acc[m][0] = __builtin_amdgcn_wmma_f32_16x16x32_bf16(
                false, ah, false, bh0.v, (short)0, acc[m][0], false, false);
            acc[m][0] = __builtin_amdgcn_wmma_f32_16x16x32_bf16(
                false, ah, false, bl0.v, (short)0, acc[m][0], false, false);
            acc[m][0] = __builtin_amdgcn_wmma_f32_16x16x32_bf16(
                false, al, false, bh0.v, (short)0, acc[m][0], false, false);

            acc[m][1] = __builtin_amdgcn_wmma_f32_16x16x32_bf16(
                false, ah, false, bh1.v, (short)0, acc[m][1], false, false);
            acc[m][1] = __builtin_amdgcn_wmma_f32_16x16x32_bf16(
                false, ah, false, bl1.v, (short)0, acc[m][1], false, false);
            acc[m][1] = __builtin_amdgcn_wmma_f32_16x16x32_bf16(
                false, al, false, bh1.v, (short)0, acc[m][1], false, false);
        }
    }

    // C/D layout (ISA 7.12.2): VGPR r -> M = r + half*8, N = nl
    #pragma unroll
    for (int m = 0; m < MTILES; ++m) {
        #pragma unroll
        for (int nt = 0; nt < 2; ++nt) {
            int tcol = t0 + wave * 32 + nt * 16 + nl;
            #pragma unroll
            for (int r = 0; r < 8; ++r) {
                int f = m * 16 + half * 8 + r;
                if (f < NFILT)
                    out[((size_t)b * NFILT + f) * T_LEN + tcol] = acc[m][nt][r];
            }
        }
    }
}

extern "C" void kernel_launch(void* const* d_in, const int* in_sizes, int n_in,
                              void* d_out, int out_size, void* d_ws, size_t ws_size,
                              hipStream_t stream) {
    const float* x    = (const float*)d_in[0];   // [32,1,32768] f32
    const float* kern = (const float*)d_in[1];   // [40,769] f32
    float* out        = (float*)d_out;           // [32,1,40,32768] f32
    bf16*  afrag      = (bf16*)d_ws;             // 2*38400 bf16 = 153.6 KB

    prep_afrag<<<(2 * AFRAG_TERM + 255) / 256, 256, 0, stream>>>(kern, afrag);

    dim3 grid(T_LEN / TILE_T, NBATCH);
    conv_wmma<<<grid, 256, 0, stream>>>(x, afrag, out);
}
